// EnhancedIPA3_27118423507360
// MI455X (gfx1250) — matrix operand order repaired and checked by reading, not compiled, not run on hardware
//
#include <hip/hip_runtime.h>
#include <hip/hip_bf16.h>

#define DEVI __device__ __forceinline__

typedef __attribute__((ext_vector_type(16))) __bf16 bf16x16;
typedef __attribute__((ext_vector_type(8)))  __bf16 bf16x8;
typedef __attribute__((ext_vector_type(8)))  float  f32x8;

constexpr int B_ = 2, N_ = 1024, CS_ = 384, H_ = 12, C_ = 16, P_ = 4, V_ = 8;
constexpr float EPS_ = 1e-8f;

// ---------------------------------------------------------------- WMMA helpers
DEVI f32x8 wmma_bf16(bf16x16 a, bf16x16 b, f32x8 c) {
  return __builtin_amdgcn_wmma_f32_16x16x32_bf16(false, a, false, b, (short)0, c,
                                                 false, false);
}

DEVI bf16x16 combine16(bf16x8 lo, bf16x8 hi) {
  bf16x16 r;
#pragma unroll
  for (int i = 0; i < 8; ++i) { r[i] = lo[i]; r[8 + i] = hi[i]; }
  return r;
}

DEVI bf16x16 combine_lo(bf16x8 lo) {
  bf16x16 r;
#pragma unroll
  for (int i = 0; i < 8; ++i) { r[i] = lo[i]; r[8 + i] = (__bf16)0.0f; }
  return r;
}

DEVI bf16x16 bzero16() {
  bf16x16 r;
#pragma unroll
  for (int i = 0; i < 16; ++i) r[i] = (__bf16)0.0f;
  return r;
}

// A fragment (16x32, K along features): row ptr has 32 contiguous bf16 feats.
// elems 0..15 = K hf*16 .. hf*16+15   (hf = lane>=16)
DEVI bf16x16 load_A32(const __bf16* row, int hf) {
  bf16x8 lo = *(const bf16x8*)(row + hf * 16);
  bf16x8 hi = *(const bf16x8*)(row + hf * 16 + 8);
  return combine16(lo, hi);
}

// B fragment: 16 contiguous bf16 along K starting at base (caller adds hf*16).
DEVI bf16x16 load_B16(const __bf16* base) {
  bf16x8 lo = *(const bf16x8*)base;
  bf16x8 hi = *(const bf16x8*)(base + 8);
  return combine16(lo, hi);
}

// ---------------------------------------------------------------- prep kernels
__global__ void __launch_bounds__(256) ipa_convert_bf16(const float* __restrict__ x,
                                                        __bf16* __restrict__ y, int n) {
  int i = blockIdx.x * blockDim.x + threadIdx.x;
  if (i < n) y[i] = (__bf16)x[i];
}

// W (K x Nout) fp32  ->  WT (Nout x K) bf16
__global__ void __launch_bounds__(256) ipa_transpose_bf16(const float* __restrict__ w,
                                                          __bf16* __restrict__ wt,
                                                          int K, int Nout) {
  int i = blockIdx.x * blockDim.x + threadIdx.x;
  if (i >= K * Nout) return;
  int k = i / Nout, c = i % Nout;
  wt[(size_t)c * K + k] = (__bf16)w[i];
}

// ------------------------------------------------------------------- bf16 GEMM
// Y[m][n] = sum_k X[m][k] * WT[n][k] + bias[n].
// One wave handles NT consecutive 16-wide N-tiles; the A fragment is loaded
// once per K-step and reused across NT WMMAs.  NT is a compile-time constant
// (dispatched on a readfirstlane'd scalar) so the inner loop is branch-free.
template <int NT>
DEVI void gemm_strip(const __bf16* __restrict__ xr, const __bf16* __restrict__ WT,
                     const float* __restrict__ bias, float* __restrict__ Y,
                     int K, int Nout, int nt0, int mt, int col, int hf) {
  const __bf16* wr[NT];
#pragma unroll
  for (int j = 0; j < NT; ++j)
    wr[j] = WT + (size_t)((nt0 + j) * 16 + col) * K + hf * 16;

  const f32x8 zacc = {0.f, 0.f, 0.f, 0.f, 0.f, 0.f, 0.f, 0.f};
  f32x8 acc[NT];
#pragma unroll
  for (int j = 0; j < NT; ++j) acc[j] = zacc;

  for (int k0 = 0; k0 < K; k0 += 32) {
    bf16x16 a = load_A32(xr + k0, hf);
#pragma unroll
    for (int j = 0; j < NT; ++j)
      acc[j] = wmma_bf16(a, load_B16(wr[j] + k0), acc[j]);
  }

  size_t rowbase = (size_t)(mt * 16 + hf * 8) * Nout;
#pragma unroll
  for (int j = 0; j < NT; ++j) {
    float bi = bias[(nt0 + j) * 16 + col];
#pragma unroll
    for (int r = 0; r < 8; ++r)
      Y[rowbase + (size_t)r * Nout + (nt0 + j) * 16 + col] = acc[j][r] + bi;
  }
}

__global__ void __launch_bounds__(128) ipa_gemm_bf16(const __bf16* __restrict__ X,
                                                     const __bf16* __restrict__ WT,
                                                     const float* __restrict__ bias,
                                                     float* __restrict__ Y,
                                                     int M, int K, int Nout) {
  (void)M;
  int lane = threadIdx.x & 31, wave = threadIdx.x >> 5;
  int mt = blockIdx.x;
  int col = lane & 15, hf = lane >> 4;
  int tiles = Nout >> 4;
  int nt0 = (blockIdx.y * 4 + wave) * 4;
  int nv = tiles - nt0;
  nv = __builtin_amdgcn_readfirstlane(nv > 4 ? 4 : nv);  // SGPR -> scalar branches
  if (nv <= 0) return;

  const __bf16* xr = X + (size_t)(mt * 16 + col) * K;
  if (nv == 4)      gemm_strip<4>(xr, WT, bias, Y, K, Nout, nt0, mt, col, hf);
  else if (nv == 3) gemm_strip<3>(xr, WT, bias, Y, K, Nout, nt0, mt, col, hf);
  else if (nv == 2) gemm_strip<2>(xr, WT, bias, Y, K, Nout, nt0, mt, col, hf);
  else              gemm_strip<1>(xr, WT, bias, Y, K, Nout, nt0, mt, col, hf);
}

// -------------------------------------------------------------- pack / geometry
// One thread per (b,n,h).  Applies relu/rot/trans/gate, folds scales into bf16
// packed operands, computes curvature + attention biases, transposes v / v_pts.
__global__ void __launch_bounds__(128) ipa_pack(
    const float* __restrict__ rawq, const float* __restrict__ rawkv,
    const float* __restrict__ rawqp, const float* __restrict__ rawkvp,
    const float* __restrict__ rawg, const float* __restrict__ rot,
    const float* __restrict__ trans, const float* __restrict__ gwp,
    __bf16* __restrict__ qpack, __bf16* __restrict__ qdpack,
    __bf16* __restrict__ kpack, __bf16* __restrict__ kdpack,
    __bf16* __restrict__ vT, __bf16* __restrict__ vpT,
    float* __restrict__ qbias, float* __restrict__ kbias) {
  int t = blockIdx.x * blockDim.x + threadIdx.x;
  if (t >= B_ * N_ * H_) return;
  int h = t % H_;
  int bn = t / H_;
  int b = bn / N_, n = bn % N_;
  size_t bhn = ((size_t)(b * H_ + h)) * N_ + n;

  float R[9], tr[3];
#pragma unroll
  for (int i = 0; i < 9; ++i) R[i] = rot[(size_t)bn * 9 + i];
#pragma unroll
  for (int i = 0; i < 3; ++i) tr[i] = trans[(size_t)bn * 3 + i];
  float gw0 = gwp[0], gw1 = gwp[1];

  // ---- q scalar + q points (gated) ----
  alignas(16) __bf16 qp[32];
  const float* qs = rawq + (size_t)bn * (H_ * C_) + h * C_;
#pragma unroll
  for (int c = 0; c < 16; ++c) qp[c] = (__bf16)(qs[c] * 0.25f);  // 1/sqrt(C)
  const float* qpr = rawqp + (size_t)bn * (H_ * P_ * 6) + h * (P_ * 6);
  const float* gg = rawg + (size_t)bn * (H_ * P_) + h * P_;
  float q2 = 0.f;
  float qd[P_][3];
#pragma unroll
  for (int p = 0; p < P_; ++p) {
    float gate = 1.f / (1.f + __expf(-gg[p]));
    float pt[6];
#pragma unroll
    for (int j = 0; j < 6; ++j) pt[j] = fmaxf(qpr[p * 6 + j], 0.f);
#pragma unroll
    for (int i = 0; i < 3; ++i) {
      float co = R[i * 3] * pt[0] + R[i * 3 + 1] * pt[1] + R[i * 3 + 2] * pt[2] + tr[i];
      float di = R[i * 3] * pt[3] + R[i * 3 + 1] * pt[4] + R[i * 3 + 2] * pt[5];
      co *= gate; di *= gate;
      q2 += co * co;
      qp[16 + p * 3 + i] = (__bf16)(co * (0.5f * gw0));  // 2*gw0/P
      qd[p][i] = di;
    }
  }
  q2 *= 0.25f;                                            // mean over P
  qp[28] = qp[29] = qp[30] = qp[31] = (__bf16)0.0f;
  {
    __bf16* dst = qpack + bhn * 32;
#pragma unroll
    for (int i = 0; i < 4; ++i) *(bf16x8*)(dst + i * 8) = *(const bf16x8*)(qp + i * 8);
  }

  // ---- k scalar + k points (ungated) ----
  alignas(16) __bf16 kp[32];
  const float* ks = rawkv + (size_t)bn * (2 * H_ * C_) + h * C_;
  const float* vs = ks + H_ * C_;
#pragma unroll
  for (int c = 0; c < 16; ++c) kp[c] = (__bf16)ks[c];
  const float* kpr = rawkvp + (size_t)bn * (H_ * (P_ + V_) * 6) + h * ((P_ + V_) * 6);
  float k2 = 0.f;
  float kd[P_][3];
#pragma unroll
  for (int p = 0; p < P_; ++p) {
    float pt[6];
#pragma unroll
    for (int j = 0; j < 6; ++j) pt[j] = fmaxf(kpr[p * 6 + j], 0.f);
#pragma unroll
    for (int i = 0; i < 3; ++i) {
      float co = R[i * 3] * pt[0] + R[i * 3 + 1] * pt[1] + R[i * 3 + 2] * pt[2] + tr[i];
      float di = R[i * 3] * pt[3] + R[i * 3 + 1] * pt[4] + R[i * 3 + 2] * pt[5];
      k2 += co * co;
      kp[16 + p * 3 + i] = (__bf16)co;
      kd[p][i] = di;
    }
  }
  k2 *= 0.25f;
  kp[28] = kp[29] = kp[30] = kp[31] = (__bf16)0.0f;
  {
    __bf16* dst = kpack + bhn * 32;
#pragma unroll
    for (int i = 0; i < 4; ++i) *(bf16x8*)(dst + i * 8) = *(const bf16x8*)(kp + i * 8);
  }

  // ---- direction packs (gw1 folded into q side) + curvature ----
  alignas(16) __bf16 qdp[16], kdp[16];
#pragma unroll
  for (int i = 12; i < 16; ++i) { qdp[i] = (__bf16)0.0f; kdp[i] = (__bf16)0.0f; }
  float curv = 0.f;
#pragma unroll
  for (int p = 0; p < P_; ++p) {
    float ax = qd[p][0], ay = qd[p][1], az = qd[p][2];
    float bx = kd[p][0], by = kd[p][1], bz = kd[p][2];
#pragma unroll
    for (int i = 0; i < 3; ++i) {
      qdp[p * 3 + i] = (__bf16)(qd[p][i] * gw1);
      kdp[p * 3 + i] = (__bf16)kd[p][i];
    }
    float cx = ay * bz - az * by, cy = az * bx - ax * bz, cz = ax * by - ay * bx;
    float cr = sqrtf(cx * cx + cy * cy + cz * cz);
    float nq = sqrtf(ax * ax + ay * ay + az * az);
    float nk = sqrtf(bx * bx + by * by + bz * bz);
    curv += cr * __builtin_amdgcn_rcpf(nq * nk + EPS_);
  }
  curv *= 0.25f;
  {
    __bf16* dq = qdpack + bhn * 16;
    __bf16* dk = kdpack + bhn * 16;
#pragma unroll
    for (int i = 0; i < 2; ++i) {
      *(bf16x8*)(dq + i * 8) = *(const bf16x8*)(qdp + i * 8);
      *(bf16x8*)(dk + i * 8) = *(const bf16x8*)(kdp + i * 8);
    }
  }
  qbias[bhn] = -gw0 * q2 - gw1 * curv;
  kbias[bhn] = -gw0 * k2;

  // ---- v transposed (b,h,c,N) ----
#pragma unroll
  for (int c = 0; c < 16; ++c)
    vT[((size_t)(b * H_ + h) * 16 + c) * N_ + n] = (__bf16)vs[c];

  // ---- v_pts transposed (b,h,48,N) ----
#pragma unroll
  for (int vi = 0; vi < V_; ++vi) {
    float pt[6];
#pragma unroll
    for (int j = 0; j < 6; ++j) pt[j] = fmaxf(kpr[(P_ + vi) * 6 + j], 0.f);
#pragma unroll
    for (int i = 0; i < 3; ++i) {
      float co = R[i * 3] * pt[0] + R[i * 3 + 1] * pt[1] + R[i * 3 + 2] * pt[2] + tr[i];
      float di = R[i * 3] * pt[3] + R[i * 3 + 1] * pt[4] + R[i * 3 + 2] * pt[5];
      vpT[((size_t)(b * H_ + h) * 48 + vi * 6 + i) * N_ + n] = (__bf16)co;
      vpT[((size_t)(b * H_ + h) * 48 + vi * 6 + 3 + i) * N_ + n] = (__bf16)di;
    }
  }
}

// --------------------------------------------------------- fused flash attention
// grid (16, B*H), block 128 (4 waves).  Wave w handles query tile blockIdx.x*4+w.
__global__ void __launch_bounds__(128) ipa_attn(
    const __bf16* __restrict__ qpack, const __bf16* __restrict__ qdpack,
    const __bf16* __restrict__ kpack, const __bf16* __restrict__ kdpack,
    const __bf16* __restrict__ vT, const __bf16* __restrict__ vpT,
    const float* __restrict__ qbias, const float* __restrict__ kbias,
    const float* __restrict__ head_weights,
    float* __restrict__ o_scalar, float* __restrict__ o_geom) {
  __shared__ __align__(16) __bf16 Pst[4][16 * 32];
  int lane = threadIdx.x & 31;
  int wave = threadIdx.x >> 5;
  int bh = blockIdx.y;
  int h = bh % H_, b = bh / H_;
  int q0 = (blockIdx.x * 4 + wave) * 16;
  int col = lane & 15, hf = lane >> 4;
  float hw = 1.f / (1.f + __expf(-head_weights[h]));
  size_t baseBH = (size_t)bh * N_;

  // query-side fragments
  bf16x16 aQ = load_A32(qpack + (baseBH + q0 + col) * 32, hf);
  bf16x16 aQD = hf ? bzero16() : load_B16(qdpack + (baseBH + q0 + col) * 16);

  float qb[8], mrun[8], lrun[8];
#pragma unroll
  for (int r = 0; r < 8; ++r) {
    qb[r] = qbias[baseBH + q0 + r + hf * 8];
    mrun[r] = -1e30f;
    lrun[r] = 0.f;
  }
  f32x8 accO = {0.f, 0.f, 0.f, 0.f, 0.f, 0.f, 0.f, 0.f};
  f32x8 accG0 = accO, accG1 = accO, accG2 = accO;
  __bf16* myP = &Pst[wave][0];

  const __bf16* vrow = vT + ((size_t)bh * 16 + col) * N_ + hf * 16;
  const __bf16* g0row = vpT + ((size_t)bh * 48 + col) * N_ + hf * 16;
  const __bf16* g1row = vpT + ((size_t)bh * 48 + 16 + col) * N_ + hf * 16;
  const __bf16* g2row = vpT + ((size_t)bh * 48 + 32 + col) * N_ + hf * 16;

  for (int kb = 0; kb < N_; kb += 32) {
    // prefetch next key block (global_prefetch_b8; speculative, dropped if OOB)
    if (kb + 32 < N_) {
      __builtin_prefetch(kpack + (baseBH + kb + 32 + col) * 32, 0, 1);
      __builtin_prefetch(vrow + kb + 32, 0, 1);
      __builtin_prefetch(g0row + kb + 32, 0, 1);
    }
    const f32x8 z = {0.f, 0.f, 0.f, 0.f, 0.f, 0.f, 0.f, 0.f};
    // scores: (q,qc) pack chained with gw1*qd·kd
    f32x8 s0 = wmma_bf16(aQ, load_B16(kpack + (baseBH + kb + col) * 32 + hf * 16), z);
    f32x8 s1 = wmma_bf16(aQ, load_B16(kpack + (baseBH + kb + 16 + col) * 32 + hf * 16), z);
    bf16x16 bkd0 = hf ? bzero16() : load_B16(kdpack + (baseBH + kb + col) * 16);
    bf16x16 bkd1 = hf ? bzero16() : load_B16(kdpack + (baseBH + kb + 16 + col) * 16);
    s0 = wmma_bf16(aQD, bkd0, s0);
    s1 = wmma_bf16(aQD, bkd1, s1);
    float kb0 = kbias[baseBH + kb + col];
    float kb1 = kbias[baseBH + kb + 16 + col];

    float p0[8], p1[8], alpha[8];
#pragma unroll
    for (int r = 0; r < 8; ++r) {
      p0[r] = (s0[r] + qb[r] + kb0) * hw;
      p1[r] = (s1[r] + qb[r] + kb1) * hw;
    }
    // online softmax (row reductions across the 16 lanes of each half)
#pragma unroll
    for (int r = 0; r < 8; ++r) {
      float tmax = fmaxf(p0[r], p1[r]);
      tmax = fmaxf(tmax, __shfl_xor(tmax, 1));
      tmax = fmaxf(tmax, __shfl_xor(tmax, 2));
      tmax = fmaxf(tmax, __shfl_xor(tmax, 4));
      tmax = fmaxf(tmax, __shfl_xor(tmax, 8));
      float mn = fmaxf(mrun[r], tmax);
      alpha[r] = __expf(mrun[r] - mn);
      mrun[r] = mn;
      p0[r] = __expf(p0[r] - mn);
      p1[r] = __expf(p1[r] - mn);
      float rs = p0[r] + p1[r];
      rs += __shfl_xor(rs, 1);
      rs += __shfl_xor(rs, 2);
      rs += __shfl_xor(rs, 4);
      rs += __shfl_xor(rs, 8);
      lrun[r] = lrun[r] * alpha[r] + rs;
    }
#pragma unroll
    for (int r = 0; r < 8; ++r) {
      accO[r] *= alpha[r];
      accG0[r] *= alpha[r];
      accG1[r] *= alpha[r];
      accG2[r] *= alpha[r];
    }
    // stage P tile (C layout -> row-major 16x32) through LDS, reload as A frag
#pragma unroll
    for (int r = 0; r < 8; ++r) {
      int m = r + hf * 8;
      myP[m * 32 + col] = (__bf16)p0[r];
      myP[m * 32 + 16 + col] = (__bf16)p1[r];
    }
    bf16x16 aP = load_A32(myP + col * 32, hf);

    accO = wmma_bf16(aP, load_B16(vrow + kb), accO);
    accG0 = wmma_bf16(aP, load_B16(g0row + kb), accG0);
    accG1 = wmma_bf16(aP, load_B16(g1row + kb), accG1);
    accG2 = wmma_bf16(aP, load_B16(g2row + kb), accG2);
  }

#pragma unroll
  for (int r = 0; r < 8; ++r) {
    float inv = __builtin_amdgcn_rcpf(lrun[r]);
    int m = q0 + r + hf * 8;
    size_t obase = ((size_t)b * N_ + m) * H_ + h;
    o_scalar[obase * 16 + col] = accO[r] * inv;
    o_geom[obase * 48 + col] = accG0[r] * inv;
    o_geom[obase * 48 + 16 + col] = accG1[r] * inv;
    o_geom[obase * 48 + 32 + col] = accG2[r] * inv;
  }
}

// ------------------------------------------------------- finalize feature rows
__global__ void __launch_bounds__(128) ipa_feats(
    const float* __restrict__ o_scalar, const float* __restrict__ o_geom,
    const float* __restrict__ rot, const float* __restrict__ trans,
    __bf16* __restrict__ feats) {
  int t = blockIdx.x * blockDim.x + threadIdx.x;
  if (t >= B_ * N_ * H_) return;
  int h = t % H_;
  int bn = t / H_;
  const float* R = rot + (size_t)bn * 9;
  const float* tr = trans + (size_t)bn * 3;
  const float* os = o_scalar + ((size_t)bn * H_ + h) * 16;
  const float* og = o_geom + ((size_t)bn * H_ + h) * 48;
  __bf16* f = feats + (size_t)bn * (H_ * (C_ + 7 * V_));
#pragma unroll
  for (int c = 0; c < 16; ++c) f[h * 16 + c] = (__bf16)os[c];
#pragma unroll
  for (int v = 0; v < V_; ++v) {
    float co[3], di[3], lc[3], ld[3];
#pragma unroll
    for (int i = 0; i < 3; ++i) { co[i] = og[v * 6 + i]; di[i] = og[v * 6 + 3 + i]; }
#pragma unroll
    for (int i = 0; i < 3; ++i) {
      lc[i] = R[i] * (co[0] - tr[0]) + R[3 + i] * (co[1] - tr[1]) + R[6 + i] * (co[2] - tr[2]);
      ld[i] = R[i] * di[0] + R[3 + i] * di[1] + R[6 + i] * di[2];
    }
    float nl = sqrtf(ld[0] * ld[0] + ld[1] * ld[1] + ld[2] * ld[2]);
    float sc = __builtin_amdgcn_rcpf(fmaxf(nl, EPS_));
    float nc = sqrtf(lc[0] * lc[0] + lc[1] * lc[1] + lc[2] * lc[2]);
    int base = H_ * C_ + h * (7 * V_) + v * 7;
#pragma unroll
    for (int i = 0; i < 3; ++i) {
      f[base + i] = (__bf16)lc[i];
      f[base + 3 + i] = (__bf16)(ld[i] * sc);
    }
    f[base + 6] = (__bf16)nc;
  }
}

// --------------------------------------------------------------------- launch
extern "C" void kernel_launch(void* const* d_in, const int* in_sizes, int n_in,
                              void* d_out, int out_size, void* d_ws, size_t ws_size,
                              hipStream_t stream) {
  (void)in_sizes; (void)n_in; (void)out_size; (void)ws_size;
  const float* s    = (const float*)d_in[0];
  const float* rot  = (const float*)d_in[1];
  const float* trns = (const float*)d_in[2];
  const float* wq   = (const float*)d_in[3];
  const float* bq   = (const float*)d_in[4];
  const float* wkv  = (const float*)d_in[5];
  const float* bkv  = (const float*)d_in[6];
  const float* wqp  = (const float*)d_in[7];
  const float* bqp  = (const float*)d_in[8];
  const float* wkvp = (const float*)d_in[9];
  const float* bkvp = (const float*)d_in[10];
  const float* wg   = (const float*)d_in[11];
  const float* bg   = (const float*)d_in[12];
  const float* gw   = (const float*)d_in[13];
  const float* hwts = (const float*)d_in[14];
  const float* wout = (const float*)d_in[15];
  const float* bout = (const float*)d_in[16];
  float* out = (float*)d_out;

  const int M = B_ * N_;                 // 2048
  const int FEAT = H_ * (C_ + 7 * V_);   // 864
  char* p = (char*)d_ws;
  auto take = [&](size_t bytes) -> char* {
    char* r = p;
    p += (bytes + 255) & ~(size_t)255;
    return r;
  };
  __bf16* sB     = (__bf16*)take((size_t)M * CS_ * 2);
  __bf16* wqT    = (__bf16*)take((size_t)192 * CS_ * 2);
  __bf16* wkvT   = (__bf16*)take((size_t)384 * CS_ * 2);
  __bf16* wqpT   = (__bf16*)take((size_t)288 * CS_ * 2);
  __bf16* wkvpT  = (__bf16*)take((size_t)864 * CS_ * 2);
  __bf16* wgT    = (__bf16*)take((size_t)48 * CS_ * 2);
  __bf16* woutT  = (__bf16*)take((size_t)CS_ * FEAT * 2);
  float* rawq    = (float*)take((size_t)M * 192 * 4);
  float* rawkv   = (float*)take((size_t)M * 384 * 4);
  float* rawqp   = (float*)take((size_t)M * 288 * 4);
  float* rawkvp  = (float*)take((size_t)M * 864 * 4);
  float* rawg    = (float*)take((size_t)M * 48 * 4);
  __bf16* qpack  = (__bf16*)take((size_t)M * H_ * 32 * 2);
  __bf16* kpack  = (__bf16*)take((size_t)M * H_ * 32 * 2);
  __bf16* qdpack = (__bf16*)take((size_t)M * H_ * 16 * 2);
  __bf16* kdpack = (__bf16*)take((size_t)M * H_ * 16 * 2);
  __bf16* vTb    = (__bf16*)take((size_t)B_ * H_ * 16 * N_ * 2);
  __bf16* vpTb   = (__bf16*)take((size_t)B_ * H_ * 48 * N_ * 2);
  float* qbias   = (float*)take((size_t)M * H_ * 4);
  float* kbias   = (float*)take((size_t)M * H_ * 4);
  float* o_sc    = (float*)take((size_t)M * H_ * 16 * 4);
  float* o_ge    = (float*)take((size_t)M * H_ * 48 * 4);
  __bf16* feats  = (__bf16*)take((size_t)M * FEAT * 2);

  // prep: convert activations, transpose+convert weights
  ipa_convert_bf16<<<(M * CS_ + 255) / 256, 256, 0, stream>>>(s, sB, M * CS_);
  ipa_transpose_bf16<<<(CS_ * 192 + 255) / 256, 256, 0, stream>>>(wq, wqT, CS_, 192);
  ipa_transpose_bf16<<<(CS_ * 384 + 255) / 256, 256, 0, stream>>>(wkv, wkvT, CS_, 384);
  ipa_transpose_bf16<<<(CS_ * 288 + 255) / 256, 256, 0, stream>>>(wqp, wqpT, CS_, 288);
  ipa_transpose_bf16<<<(CS_ * 864 + 255) / 256, 256, 0, stream>>>(wkvp, wkvpT, CS_, 864);
  ipa_transpose_bf16<<<(CS_ * 48 + 255) / 256, 256, 0, stream>>>(wg, wgT, CS_, 48);
  ipa_transpose_bf16<<<(FEAT * CS_ + 255) / 256, 256, 0, stream>>>(wout, woutT, FEAT, CS_);

  // GEMMs: each wave covers up to 4 N-tiles (16x64); block of 4 waves -> 256 cols
  auto gemm = [&](const __bf16* X, const __bf16* WT, const float* bias, float* Y,
                  int K, int Nout) {
    int tiles = Nout / 16;
    dim3 grid(M / 16, (tiles + 15) / 16);
    ipa_gemm_bf16<<<grid, 128, 0, stream>>>(X, WT, bias, Y, M, K, Nout);
  };
  gemm(sB, wqT, bq, rawq, CS_, 192);
  gemm(sB, wkvT, bkv, rawkv, CS_, 384);
  gemm(sB, wqpT, bqp, rawqp, CS_, 288);
  gemm(sB, wkvpT, bkvp, rawkvp, CS_, 864);
  gemm(sB, wgT, bg, rawg, CS_, 48);

  // pack + geometry
  int nthr = B_ * N_ * H_;
  ipa_pack<<<(nthr + 127) / 128, 128, 0, stream>>>(
      rawq, rawkv, rawqp, rawkvp, rawg, rot, trns, gw, qpack, qdpack, kpack,
      kdpack, vTb, vpTb, qbias, kbias);

  // fused flash attention
  ipa_attn<<<dim3(N_ / 16 / 4, B_ * H_), 128, 0, stream>>>(
      qpack, qdpack, kpack, kdpack, vTb, vpTb, qbias, kbias, hwts, o_sc, o_ge);

  // finalize features + output GEMM
  ipa_feats<<<(nthr + 127) / 128, 128, 0, stream>>>(o_sc, o_ge, rot, trns, feats);
  gemm(feats, woutT, bout, out, FEAT, CS_);
}
